// Gru_cond_layer_53644141527200
// MI455X (gfx1250) — matrix operand order repaired and checked by reading, not compiled, not run on hardware
//
#include <hip/hip_runtime.h>
#include <hip/hip_bf16.h>
#include <math.h>

// ---------------- problem dimensions ----------------
#define T32   32
#define B8    8
#define M256  256
#define N256  256
#define D684  684
#define DA512 512
#define H12   12
#define W48   48
#define HW    576          // H*W
#define HWB   4608         // H*W*B
#define KPD   704          // D padded to multiple of 32
#define K2M   512          // 2*M

// ---------------- WMMA types ----------------
typedef __attribute__((ext_vector_type(16))) __bf16 v16bf;
typedef __attribute__((ext_vector_type(8)))  __bf16 v8bf;
typedef __attribute__((ext_vector_type(8)))  float  v8f;

union ABfrag { v16bf v; v8bf h[2]; __bf16 e[16]; };
union CDfrag { v8f   v; float  f[8];  };

static __device__ __forceinline__ float sigmoidf_(float x) {
    return 1.0f / (1.0f + __expf(-x));
}

// =====================================================================
// Generic GEMM: out[i,n] = sum_k A_bf16[i*lda+k] * Wb_bf16[n*ldw+k] (+bias)
// One wave per 16x16 output tile, K in chunks of 32 via
// v_wmma_f32_16x16x32_bf16.  Both operands are pre-staged bf16, rows
// padded (zero) to 16-multiples, K zero-padded to KP (mult of 32).
// Per ISA 7.12.2 the per-lane slices of a K=32 chunk are contiguous:
//   A lane(m,half):  [kk+8*half .. +7] and [kk+16+8*half .. +7]
//   B lane(n,half):  [kk+8*half .. +7] and [kk+16+8*half .. +7]
// -> four 16-byte vector loads + one WMMA per chunk, no branches.
// =====================================================================
template<int KP>
__global__ void gemm_wmma(const __bf16* __restrict__ A, int lda,
                          const __bf16* __restrict__ Wb, int ldw,
                          const float* __restrict__ bias,
                          float* __restrict__ out, int ldo,
                          int Mrows, int tilesN, int nTiles)
{
    int wave = (int)((blockIdx.x * blockDim.x + threadIdx.x) >> 5);
    if (wave >= nTiles) return;              // whole-wave exit: EXEC all-1s for WMMA
    int lane = threadIdx.x & 31;
    int tM = wave / tilesN, tN = wave - tM * tilesN;
    int i0 = tM << 4, n0 = tN << 4;
    int m = lane & 15, half = lane >> 4;

    const v8bf* Ar = (const v8bf*)(A + (size_t)(i0 + m) * lda);
    const v8bf* Wr = (const v8bf*)(Wb + (size_t)(n0 + m) * ldw);

    CDfrag acc;
#pragma unroll
    for (int r = 0; r < 8; ++r) acc.f[r] = 0.0f;

#pragma unroll 4
    for (int c = 0; c < KP / 32; ++c) {
        ABfrag a, b;
        a.h[0] = Ar[4 * c + half];           // global_load_b128
        a.h[1] = Ar[4 * c + 2 + half];
        b.h[0] = Wr[4 * c + half];
        b.h[1] = Wr[4 * c + 2 + half];
        acc.v = __builtin_amdgcn_wmma_f32_16x16x32_bf16(
            false, a.v, false, b.v, (short)0, acc.v, false, false);
    }

    int col = n0 + m;
    float vb = bias ? bias[col] : 0.0f;
#pragma unroll
    for (int r = 0; r < 8; ++r) {
        int row = i0 + r + half * 8;        // C/D layout: VGPR r -> rows r, r+8
        if (row < Mrows)
            out[(size_t)row * ldo + col] = acc.f[r] + vb;
    }
}

// ---------------- init / staging kernels ----------------
__global__ void zero_f32_kernel(float* p, int n) {
    int i = blockIdx.x * blockDim.x + threadIdx.x;
    if (i < n) p[i] = 0.0f;
}
__global__ void zero_bf16_kernel(__bf16* p, int n) {
    int i = blockIdx.x * blockDim.x + threadIdx.x;
    if (i < n) p[i] = (__bf16)0.0f;
}
__global__ void inith_kernel(const float* __restrict__ init_state,
                             float* hcur, __bf16* hA) {
    int i = blockIdx.x * blockDim.x + threadIdx.x;
    if (i >= B8 * N256) return;
    float v = init_state[i];
    hcur[i] = v;
    hA[i] = (__bf16)v;                      // rows 8..15 pre-zeroed
}

// generic fp32 -> bf16 weight staging with K zero-padding: out[n*Kp+k]
__global__ void wconv_kernel(const float* __restrict__ W, __bf16* __restrict__ out,
                             int N, int K, int Kp) {
    int total = N * Kp;
    for (int idx = blockIdx.x * blockDim.x + threadIdx.x; idx < total;
         idx += gridDim.x * blockDim.x) {
        int n = idx / Kp, k = idx - n * Kp;
        out[idx] = (__bf16)((k < K) ? W[(size_t)n * K + k] : 0.0f);
    }
}

// context[b,d,h,w] -> ctxA[(hw*B+b)*KPD + d]  (bf16, K zero-padded)
__global__ void ctxa_kernel(const float* __restrict__ ctx, __bf16* __restrict__ A) {
    const size_t total = (size_t)HWB * KPD;
    for (size_t idx = (size_t)blockIdx.x * blockDim.x + threadIdx.x;
         idx < total; idx += (size_t)gridDim.x * blockDim.x) {
        int i = (int)(idx / KPD);
        int d = (int)(idx - (size_t)i * KPD);
        int b = i & 7, hw = i >> 3;
        float v = (d < D684) ? ctx[((size_t)b * D684 + d) * HW + hw] : 0.0f;
        A[idx] = (__bf16)v;
    }
}

// emb = concat(remb, re_emb) -> embA[(t*B+b)*512 + k]
__global__ void emba_kernel(const float* __restrict__ remb,
                            const float* __restrict__ re_emb,
                            __bf16* __restrict__ A) {
    int idx = blockIdx.x * blockDim.x + threadIdx.x;
    if (idx >= T32 * B8 * K2M) return;
    int i = idx >> 9, k = idx & 511;
    float v = (k < M256) ? remb[i * M256 + k] : re_emb[i * M256 + (k - M256)];
    A[idx] = (__bf16)v;
}

// Ueff[a,j] = sum_c UfC[a,c] * QC_w[c,0,j]  (folds conv + 1x1 projection)
__global__ void ueff_kernel(const float* __restrict__ UfC_w,
                            const float* __restrict__ QC_w,
                            float* __restrict__ Ueff) {
    int idx = blockIdx.x * blockDim.x + threadIdx.x;
    if (idx >= DA512 * 9) return;
    int a = idx / 9, j = idx - a * 9;
    float s = 0.0f;
    for (int c = 0; c < 512; ++c) s += UfC_w[a * 512 + c] * QC_w[c * 9 + j];
    Ueff[idx] = s;
}

// ---------------- per-step kernels ----------------
__global__ void ctp_kernel(const float* __restrict__ ctCbuf,
                           const int* __restrict__ rp,
                           float* __restrict__ out_ctP, int t) {
    int i = blockIdx.x * blockDim.x + threadIdx.x;
    if (i >= B8 * D684) return;
    int b = i / D684, d = i - b * D684;
    int pos = rp[t * B8 + b];
    out_ctP[(size_t)(t * B8 + b) * D684 + d] =
        ctCbuf[(size_t)pos * (B8 * D684) + i];
}

__global__ void h0_kernel(const float* __restrict__ tmpz, const float* __restrict__ tmpr,
                          const float* __restrict__ tmph,
                          const float* __restrict__ sbz, const float* __restrict__ sbr,
                          const float* __restrict__ sbh,
                          const float* __restrict__ lymask,
                          const float* __restrict__ hprev,
                          float* __restrict__ h0, __bf16* __restrict__ h0A, int t) {
    int i = blockIdx.x * blockDim.x + threadIdx.x;
    if (i >= B8 * N256) return;
    int b = i >> 8;
    int sb = t * (B8 * N256) + i;
    float h_ = hprev[i];
    float z0 = sigmoidf_(tmpz[i] + sbz[sb]);
    float r0 = sigmoidf_(tmpr[i] + sbr[sb]);
    float hp = tanhf(tmph[i] * r0 + sbh[sb]);
    float h  = z0 * h_ + (1.0f - z0) * hp;
    float lm = lymask[t * B8 + b];
    h = lm * h + (1.0f - lm) * h_;
    h0[i]  = h;
    h0A[i] = (__bf16)h;                     // rows 8..15 stay zero
}

// e[pos] = vaC . tanh(pctx[pos,:] + query[b,:] + Ueff*ap_window + UfC_b) + vaC_b
// one wave per (h,w,b) position
__global__ void score_kernel(const float* __restrict__ pctx,
                             const float* __restrict__ query,
                             const float* __restrict__ Ueff,
                             const float* __restrict__ ufcb,
                             const float* __restrict__ ap,
                             const float* __restrict__ vaC_w,
                             const float* __restrict__ vaC_b,
                             float* __restrict__ e) {
    int wid  = (int)((blockIdx.x * blockDim.x + threadIdx.x) >> 5);
    int lane = threadIdx.x & 31;
    if (wid >= HWB) return;
    int hw = wid >> 3, b = wid & 7;
    int h = hw / W48, w = hw - h * W48;
    float apw[9];
#pragma unroll
    for (int dy = 0; dy < 3; ++dy)
#pragma unroll
        for (int dx = 0; dx < 3; ++dx) {
            int hh = h + dy - 1, ww = w + dx - 1;
            apw[dy * 3 + dx] = (hh >= 0 && hh < H12 && ww >= 0 && ww < W48)
                                   ? ap[b * HW + hh * W48 + ww] : 0.0f;
        }
    const float* pc = pctx + (size_t)wid * DA512;
    const float* q  = query + b * DA512;
    float acc = 0.0f;
    for (int a = lane; a < DA512; a += 32) {
        const float* ue = Ueff + a * 9;
        float cov = 0.0f;
#pragma unroll
        for (int j = 0; j < 9; ++j) cov += ue[j] * apw[j];
        acc += vaC_w[a] * tanhf(pc[a] + q[a] + cov + ufcb[a]);
    }
    for (int off = 16; off; off >>= 1) acc += __shfl_xor(acc, off, 32);
    if (lane == 0) e[wid] = acc + vaC_b[0];
}

__global__ void max_kernel(const float* __restrict__ e, float* __restrict__ maxv) {
    __shared__ float sm[256];
    float m = -3.0e38f;
    for (int i = threadIdx.x; i < HWB; i += 256) m = fmaxf(m, e[i]);
    sm[threadIdx.x] = m;
    __syncthreads();
    for (int s = 128; s; s >>= 1) {
        if (threadIdx.x < s) sm[threadIdx.x] = fmaxf(sm[threadIdx.x], sm[threadIdx.x + s]);
        __syncthreads();
    }
    if (threadIdx.x == 0) maxv[0] = sm[0];
}

__global__ void exp_kernel(const float* __restrict__ e, const float* __restrict__ maxv,
                           const float* __restrict__ cmask, float* __restrict__ abuf) {
    int i = blockIdx.x * blockDim.x + threadIdx.x;
    if (i >= HWB) return;
    int hw = i >> 3, b = i & 7;
    abuf[i] = __expf(e[i] - maxv[0]) * cmask[b * HW + hw];
}

__global__ void sum_kernel(const float* __restrict__ abuf, float* __restrict__ sums) {
    int b = blockIdx.x;                     // 8 blocks
    __shared__ float sm[256];
    float s = 0.0f;
    for (int hw = threadIdx.x; hw < HW; hw += 256) s += abuf[hw * B8 + b];
    sm[threadIdx.x] = s;
    __syncthreads();
    for (int k = 128; k; k >>= 1) {
        if (threadIdx.x < k) sm[threadIdx.x] += sm[threadIdx.x + k];
        __syncthreads();
    }
    if (threadIdx.x == 0) sums[b] = sm[0];
}

__global__ void fin_kernel(float* __restrict__ abuf, const float* __restrict__ sums,
                           float* __restrict__ ap,
                           float* __restrict__ out_alpha, float* __restrict__ out_apast,
                           int t) {
    int i = blockIdx.x * blockDim.x + threadIdx.x;
    if (i >= HWB) return;
    int hw = i >> 3, b = i & 7;
    float a = abuf[i] / sums[b] + 1e-10f;
    abuf[i] = a;
    int oi = (t * B8 + b) * HW + hw;
    out_alpha[oi] = a;
    float apn = ap[b * HW + hw] + a;
    ap[b * HW + hw] = apn;
    out_apast[oi] = apn;
}

// ctC[b,d] = sum_hw context[b,d,hw] * alpha[hw,b]
__global__ void ctc_kernel(const float* __restrict__ context,
                           const float* __restrict__ abuf,
                           float* __restrict__ ctCbuf, __bf16* __restrict__ ctCA,
                           float* __restrict__ out_ctC, int t) {
    int i = blockIdx.x * blockDim.x + threadIdx.x;
    if (i >= B8 * D684) return;
    int b = i / D684, d = i - b * D684;
    const float* cp = context + (size_t)i * HW;
    float s = 0.0f;
    for (int hw = 0; hw < HW; ++hw) {
        if ((hw & 31) == 0) __builtin_prefetch(cp + hw + 64, 0, 1); // global_prefetch_b8
        s += cp[hw] * abuf[hw * B8 + b];
    }
    ctCbuf[(size_t)(t + 1) * (B8 * D684) + i] = s;
    ctCA[b * KPD + d] = (__bf16)s;          // pad region pre-zeroed
    out_ctC[(size_t)(t * B8 + b) * D684 + d] = s;
}

__global__ void h2_kernel(const float* __restrict__ tmp2z, const float* __restrict__ tmp2r,
                          const float* __restrict__ tmp2h,
                          const float* __restrict__ tmpcz, const float* __restrict__ tmpcr,
                          const float* __restrict__ tmpch,
                          const float* __restrict__ lymask, const float* __restrict__ h0,
                          float* __restrict__ hcur, __bf16* __restrict__ hA,
                          float* __restrict__ out_h2, int t) {
    int i = blockIdx.x * blockDim.x + threadIdx.x;
    if (i >= B8 * N256) return;
    int b = i >> 8;
    float h0v = h0[i];
    float z2 = sigmoidf_(tmp2z[i] + tmpcz[i]);
    float r2 = sigmoidf_(tmp2r[i] + tmpcr[i]);
    float hp = tanhf(tmp2h[i] * r2 + tmpch[i]);
    float h  = z2 * h0v + (1.0f - z2) * hp;
    float lm = lymask[t * B8 + b];
    h = lm * h + (1.0f - lm) * h0v;
    hcur[i] = h;
    hA[i]   = (__bf16)h;
    out_h2[(size_t)t * (B8 * N256) + i] = h;
}

// ---------------- workspace layout ----------------
static constexpr size_t AL(size_t x) { return (x + 255) & ~(size_t)255; }
static constexpr size_t OFF_CTXA  = 0;
static constexpr size_t OFF_EMBA  = AL(OFF_CTXA  + (size_t)HWB * KPD * 2);
static constexpr size_t OFF_PCTX  = AL(OFF_EMBA  + (size_t)T32 * B8 * K2M * 2);
static constexpr size_t OFF_SBZ   = AL(OFF_PCTX  + (size_t)HWB * DA512 * 4);
static constexpr size_t OFF_SBR   = AL(OFF_SBZ   + (size_t)T32 * B8 * N256 * 4);
static constexpr size_t OFF_SBH   = AL(OFF_SBR   + (size_t)T32 * B8 * N256 * 4);
static constexpr size_t OFF_UEFF  = AL(OFF_SBH   + (size_t)T32 * B8 * N256 * 4);
static constexpr size_t OFF_HA    = AL(OFF_UEFF  + (size_t)DA512 * 9 * 4);
static constexpr size_t OFF_H0A   = AL(OFF_HA    + (size_t)16 * N256 * 2);
static constexpr size_t OFF_CTCA  = AL(OFF_H0A   + (size_t)16 * N256 * 2);
static constexpr size_t OFF_HCUR  = AL(OFF_CTCA  + (size_t)16 * KPD * 2);
static constexpr size_t OFF_H0    = AL(OFF_HCUR  + (size_t)B8 * N256 * 4);
static constexpr size_t OFF_TMP   = AL(OFF_H0    + (size_t)B8 * N256 * 4);   // 9 x B*N
static constexpr size_t OFF_QUERY = AL(OFF_TMP   + (size_t)9 * B8 * N256 * 4);
static constexpr size_t OFF_E     = AL(OFF_QUERY + (size_t)B8 * DA512 * 4);
static constexpr size_t OFF_ABUF  = AL(OFF_E     + (size_t)HWB * 4);
static constexpr size_t OFF_AP    = AL(OFF_ABUF  + (size_t)HWB * 4);
static constexpr size_t OFF_SUMS  = AL(OFF_AP    + (size_t)HWB * 4);
static constexpr size_t OFF_MAXV  = AL(OFF_SUMS  + 64);
static constexpr size_t OFF_CTCB  = AL(OFF_MAXV  + 64);                      // (T+1)*B*D f32
// bf16 staged weights (zero-padded K)
static constexpr size_t OFF_WUHZ0 = AL(OFF_CTCB  + (size_t)(T32 + 1) * B8 * D684 * 4);
static constexpr size_t OFF_WUHR0 = AL(OFF_WUHZ0 + (size_t)N256 * N256 * 2);
static constexpr size_t OFF_WUHH0 = AL(OFF_WUHR0 + (size_t)N256 * N256 * 2);
static constexpr size_t OFF_WUHZ2 = AL(OFF_WUHH0 + (size_t)N256 * N256 * 2);
static constexpr size_t OFF_WUHR2 = AL(OFF_WUHZ2 + (size_t)N256 * N256 * 2);
static constexpr size_t OFF_WUHH2 = AL(OFF_WUHR2 + (size_t)N256 * N256 * 2);
static constexpr size_t OFF_WWAC  = AL(OFF_WUHH2 + (size_t)N256 * N256 * 2);
static constexpr size_t OFF_WWCZ  = AL(OFF_WWAC  + (size_t)DA512 * N256 * 2);
static constexpr size_t OFF_WWCR  = AL(OFF_WWCZ  + (size_t)N256 * KPD * 2);
static constexpr size_t OFF_WWCH  = AL(OFF_WWCR  + (size_t)N256 * KPD * 2);
static constexpr size_t OFF_WUAC  = AL(OFF_WWCH  + (size_t)N256 * KPD * 2);
static constexpr size_t OFF_WWYZ  = AL(OFF_WUAC  + (size_t)DA512 * KPD * 2);
static constexpr size_t OFF_WWYR  = AL(OFF_WWYZ  + (size_t)N256 * K2M * 2);
static constexpr size_t OFF_WWYH  = AL(OFF_WWYR  + (size_t)N256 * K2M * 2);

// output offsets (elements)
static constexpr size_t O_H2  = 0;
static constexpr size_t O_CTC = O_H2  + (size_t)T32 * B8 * N256;
static constexpr size_t O_CTP = O_CTC + (size_t)T32 * B8 * D684;
static constexpr size_t O_AL  = O_CTP + (size_t)T32 * B8 * D684;
static constexpr size_t O_AP  = O_AL  + (size_t)T32 * B8 * HW;

template<int KP>
static inline void launch_gemm(hipStream_t s, const __bf16* A, int lda,
                               const __bf16* Wb, int ldw, const float* bias,
                               float* out, int ldo,
                               int Mrows, int MrowsPad, int Ncols) {
    int tilesN = Ncols / 16, nT = (MrowsPad / 16) * tilesN;
    int blocks = (nT * 32 + 255) / 256;
    gemm_wmma<KP><<<blocks, 256, 0, s>>>(A, lda, Wb, ldw, bias, out, ldo,
                                         Mrows, tilesN, nT);
}

static inline void launch_wconv(hipStream_t s, const float* W, __bf16* out,
                                int N, int K, int Kp) {
    int total = N * Kp;
    int blocks = (total + 255) / 256;
    if (blocks > 1024) blocks = 1024;
    wconv_kernel<<<blocks, 256, 0, s>>>(W, out, N, K, Kp);
}

extern "C" void kernel_launch(void* const* d_in, const int* in_sizes, int n_in,
                              void* d_out, int out_size, void* d_ws, size_t ws_size,
                              hipStream_t stream) {
    const float* rembedding   = (const float*)d_in[0];
    const float* re_embedding = (const float*)d_in[1];
    const float* ly_mask      = (const float*)d_in[2];
    const float* context      = (const float*)d_in[3];
    const float* context_mask = (const float*)d_in[4];
    const float* init_state   = (const float*)d_in[5];
    const float* Wyz_w = (const float*)d_in[6];  const float* Wyz_b = (const float*)d_in[7];
    const float* Wyr_w = (const float*)d_in[8];  const float* Wyr_b = (const float*)d_in[9];
    const float* Wyh_w = (const float*)d_in[10]; const float* Wyh_b = (const float*)d_in[11];
    const float* Uhz0_w = (const float*)d_in[12];
    const float* Uhr0_w = (const float*)d_in[13];
    const float* Uhh0_w = (const float*)d_in[14];
    const float* UaC_w = (const float*)d_in[15]; const float* UaC_b = (const float*)d_in[16];
    const float* WaC_w = (const float*)d_in[17];
    const float* QC_w  = (const float*)d_in[18];
    const float* UfC_w = (const float*)d_in[19]; const float* UfC_b = (const float*)d_in[20];
    const float* vaC_w = (const float*)d_in[21]; const float* vaC_b = (const float*)d_in[22];
    const float* Uhz2_w = (const float*)d_in[23]; const float* Uhz2_b = (const float*)d_in[24];
    const float* Uhr2_w = (const float*)d_in[25]; const float* Uhr2_b = (const float*)d_in[26];
    const float* Uhh2_w = (const float*)d_in[27]; const float* Uhh2_b = (const float*)d_in[28];
    const float* Wcz_w = (const float*)d_in[29];
    const float* Wcr_w = (const float*)d_in[30];
    const float* Wch_w = (const float*)d_in[31];
    const int*   rp    = (const int*)d_in[32];

    char* ws = (char*)d_ws;
    __bf16* ctxA  = (__bf16*)(ws + OFF_CTXA);
    __bf16* embA  = (__bf16*)(ws + OFF_EMBA);
    float*  pctx  = (float*)(ws + OFF_PCTX);
    float*  sbz   = (float*)(ws + OFF_SBZ);
    float*  sbr   = (float*)(ws + OFF_SBR);
    float*  sbh   = (float*)(ws + OFF_SBH);
    float*  Ueff  = (float*)(ws + OFF_UEFF);
    __bf16* hA    = (__bf16*)(ws + OFF_HA);
    __bf16* h0A   = (__bf16*)(ws + OFF_H0A);
    __bf16* ctCA  = (__bf16*)(ws + OFF_CTCA);
    float*  hcur  = (float*)(ws + OFF_HCUR);
    float*  h0    = (float*)(ws + OFF_H0);
    float*  tmp   = (float*)(ws + OFF_TMP);   // 9 slabs of B*N
    float*  tmpz  = tmp + 0 * 2048, *tmpr  = tmp + 1 * 2048, *tmph  = tmp + 2 * 2048;
    float*  tmp2z = tmp + 3 * 2048, *tmp2r = tmp + 4 * 2048, *tmp2h = tmp + 5 * 2048;
    float*  tmpcz = tmp + 6 * 2048, *tmpcr = tmp + 7 * 2048, *tmpch = tmp + 8 * 2048;
    float*  query = (float*)(ws + OFF_QUERY);
    float*  e     = (float*)(ws + OFF_E);
    float*  abuf  = (float*)(ws + OFF_ABUF);
    float*  ap    = (float*)(ws + OFF_AP);
    float*  sums  = (float*)(ws + OFF_SUMS);
    float*  maxv  = (float*)(ws + OFF_MAXV);
    float*  ctCbuf = (float*)(ws + OFF_CTCB);
    __bf16* wUhz0 = (__bf16*)(ws + OFF_WUHZ0);
    __bf16* wUhr0 = (__bf16*)(ws + OFF_WUHR0);
    __bf16* wUhh0 = (__bf16*)(ws + OFF_WUHH0);
    __bf16* wUhz2 = (__bf16*)(ws + OFF_WUHZ2);
    __bf16* wUhr2 = (__bf16*)(ws + OFF_WUHR2);
    __bf16* wUhh2 = (__bf16*)(ws + OFF_WUHH2);
    __bf16* wWaC  = (__bf16*)(ws + OFF_WWAC);
    __bf16* wWcz  = (__bf16*)(ws + OFF_WWCZ);
    __bf16* wWcr  = (__bf16*)(ws + OFF_WWCR);
    __bf16* wWch  = (__bf16*)(ws + OFF_WWCH);
    __bf16* wUaC  = (__bf16*)(ws + OFF_WUAC);
    __bf16* wWyz  = (__bf16*)(ws + OFF_WWYZ);
    __bf16* wWyr  = (__bf16*)(ws + OFF_WWYR);
    __bf16* wWyh  = (__bf16*)(ws + OFF_WWYH);

    float* out = (float*)d_out;
    float* out_h2  = out + O_H2;
    float* out_ctC = out + O_CTC;
    float* out_ctP = out + O_CTP;
    float* out_al  = out + O_AL;
    float* out_ap  = out + O_AP;

    // ---------- per-call init (graph-replay safe, no atomics anywhere) ----------
    {
        int n = (T32 + 1) * B8 * D684;
        zero_f32_kernel<<<(n + 255) / 256, 256, 0, stream>>>(ctCbuf, n);
        zero_f32_kernel<<<(HWB + 255) / 256, 256, 0, stream>>>(ap, HWB);
        zero_bf16_kernel<<<(16 * N256 + 255) / 256, 256, 0, stream>>>(hA, 16 * N256);
        zero_bf16_kernel<<<(16 * N256 + 255) / 256, 256, 0, stream>>>(h0A, 16 * N256);
        zero_bf16_kernel<<<(16 * KPD + 255) / 256, 256, 0, stream>>>(ctCA, 16 * KPD);
        inith_kernel<<<(B8 * N256 + 255) / 256, 256, 0, stream>>>(init_state, hcur, hA);
    }

    // ---------- stage weights as padded bf16 (once per call) ----------
    launch_wconv(stream, Uhz0_w, wUhz0, N256, N256, N256);
    launch_wconv(stream, Uhr0_w, wUhr0, N256, N256, N256);
    launch_wconv(stream, Uhh0_w, wUhh0, N256, N256, N256);
    launch_wconv(stream, Uhz2_w, wUhz2, N256, N256, N256);
    launch_wconv(stream, Uhr2_w, wUhr2, N256, N256, N256);
    launch_wconv(stream, Uhh2_w, wUhh2, N256, N256, N256);
    launch_wconv(stream, WaC_w,  wWaC,  DA512, N256, N256);
    launch_wconv(stream, Wcz_w,  wWcz,  N256, D684, KPD);
    launch_wconv(stream, Wcr_w,  wWcr,  N256, D684, KPD);
    launch_wconv(stream, Wch_w,  wWch,  N256, D684, KPD);
    launch_wconv(stream, UaC_w,  wUaC,  DA512, D684, KPD);
    launch_wconv(stream, Wyz_w,  wWyz,  N256, K2M, K2M);
    launch_wconv(stream, Wyr_w,  wWyr,  N256, K2M, K2M);
    launch_wconv(stream, Wyh_w,  wWyh,  N256, K2M, K2M);

    // ---------- precompute (once): staging + big GEMMs ----------
    ctxa_kernel<<<4096, 256, 0, stream>>>(context, ctxA);
    emba_kernel<<<(T32 * B8 * K2M + 255) / 256, 256, 0, stream>>>(rembedding, re_embedding, embA);
    ueff_kernel<<<(DA512 * 9 + 255) / 256, 256, 0, stream>>>(UfC_w, QC_w, Ueff);

    // pctx[hwb, a] = context . UaC^T + b    (4608 x 512 x 684)
    launch_gemm<KPD>(stream, ctxA, KPD, wUaC, KPD, UaC_b, pctx, DA512, HWB, HWB, DA512);
    // sb* = emb @ Wy*^T + b                 (256 x 256 x 512)
    launch_gemm<K2M>(stream, embA, K2M, wWyz, K2M, Wyz_b, sbz, N256, 256, 256, N256);
    launch_gemm<K2M>(stream, embA, K2M, wWyr, K2M, Wyr_b, sbr, N256, 256, 256, N256);
    launch_gemm<K2M>(stream, embA, K2M, wWyh, K2M, Wyh_b, sbh, N256, 256, 256, N256);

    // ---------- sequential scan ----------
    for (int t = 0; t < T32; ++t) {
        // parent-context gather (before this step writes slot t+1)
        ctp_kernel<<<(B8 * D684 + 255) / 256, 256, 0, stream>>>(ctCbuf, rp, out_ctP, t);

        // GRU cell 1: h_ @ U{z,r,h}0^T  (8x256x256 each, WMMA)
        launch_gemm<N256>(stream, hA, N256, wUhz0, N256, nullptr, tmpz, N256, B8, 16, N256);
        launch_gemm<N256>(stream, hA, N256, wUhr0, N256, nullptr, tmpr, N256, B8, 16, N256);
        launch_gemm<N256>(stream, hA, N256, wUhh0, N256, nullptr, tmph, N256, B8, 16, N256);
        h0_kernel<<<(B8 * N256 + 255) / 256, 256, 0, stream>>>(
            tmpz, tmpr, tmph, sbz, sbr, sbh, ly_mask, hcur, h0, h0A, t);

        // query = h0 @ WaC^T (8x512x256, WMMA)
        launch_gemm<N256>(stream, h0A, N256, wWaC, N256, nullptr, query, DA512, B8, 16, DA512);

        // fused coverage-attention score (conv folded into Ueff)
        score_kernel<<<(HWB * 32 + 255) / 256, 256, 0, stream>>>(
            pctx, query, Ueff, UfC_b, ap, vaC_w, vaC_b, e);
        max_kernel<<<1, 256, 0, stream>>>(e, maxv);
        exp_kernel<<<(HWB + 255) / 256, 256, 0, stream>>>(e, maxv, context_mask, abuf);
        sum_kernel<<<B8, 256, 0, stream>>>(abuf, sums);
        fin_kernel<<<(HWB + 255) / 256, 256, 0, stream>>>(abuf, sums, ap, out_al, out_ap, t);

        // ctC = context . alpha  (+ stage bf16, ring buffer, output)
        ctc_kernel<<<(B8 * D684 + 255) / 256, 256, 0, stream>>>(
            context, abuf, ctCbuf, ctCA, out_ctC, t);

        // GRU cell 2: h0 @ U*2^T (+bias) and ctC @ Wc*^T  (WMMA)
        launch_gemm<N256>(stream, h0A, N256, wUhz2, N256, Uhz2_b, tmp2z, N256, B8, 16, N256);
        launch_gemm<N256>(stream, h0A, N256, wUhr2, N256, Uhr2_b, tmp2r, N256, B8, 16, N256);
        launch_gemm<N256>(stream, h0A, N256, wUhh2, N256, Uhh2_b, tmp2h, N256, B8, 16, N256);
        launch_gemm<KPD>(stream, ctCA, KPD, wWcz, KPD, nullptr, tmpcz, N256, B8, 16, N256);
        launch_gemm<KPD>(stream, ctCA, KPD, wWcr, KPD, nullptr, tmpcr, N256, B8, 16, N256);
        launch_gemm<KPD>(stream, ctCA, KPD, wWch, KPD, nullptr, tmpch, N256, B8, 16, N256);

        h2_kernel<<<(B8 * N256 + 255) / 256, 256, 0, stream>>>(
            tmp2z, tmp2r, tmp2h, tmpcz, tmpcr, tmpch, ly_mask, h0,
            hcur, hA, out_h2, t);
    }
}
